// PCN_11098195493090
// MI455X (gfx1250) — compile-verified, matrix operand order checked by reading, not run on hardware
//
#include <hip/hip_runtime.h>
#include <hip/hip_bf16.h>

typedef __bf16 bf16_t;
typedef bf16_t v16bf __attribute__((ext_vector_type(16)));
typedef float  v8f   __attribute__((ext_vector_type(8)));

#define PCN_B     8192
#define PCN_DIN   784
#define PCN_DINP  800     // K padded to multiple of 32
#define PCN_D1    256
#define PCN_D2    1024
#define PCN_D3    256
#define PCN_DOUT  10
#define PCN_STEPS 20
#define PCN_LR    0.1f
#define PCN_GAMMA 0.1f
#define MT        32      // batch rows per block

// LDS row strides padded by +8 bf16 (16B) so successive rows land 4 banks
// apart: 16 rows span all 64 LDS banks -> conflict-free ds_load_b128.
#define LD1 (PCN_D1 + 8)  // 264
#define LD2 (PCN_D2 + 8)  // 1032
#define LD3 (PCN_D3 + 8)  // 264
#define LD4 40            // eps4: 32 + 8

union FragU { v16bf v; uint4 u[2]; bf16_t e[16]; };

__device__ __forceinline__ float sigmoidf_(float v) {
  return 1.0f / (1.0f + __expf(-v));
}

// C(16x16,f32) += A(16xK bf16, row-major LDS, row stride lda)
//              @ B(KxN bf16, column-major global: column n at B + n*ldb)
// Fragment runs match the CDNA5 16-bit A-matrix layout:
//   lanes 0-15 : VGPR0-3 = K k..k+7,   VGPR4-7 = K k+16..k+23
//   lanes 16-31: VGPR0-3 = K k+8..k+15,VGPR4-7 = K k+24..k+31
// Dual accumulator chains (even/odd K-slice) so the scheduler can issue
// back-to-back WMMAs without a serial RAW on the accumulator.
__device__ __forceinline__ v8f wmma_AB(const bf16_t* A, int lda,
                                       const bf16_t* B, int ldb,
                                       int K, v8f c) {
  const int lane = threadIdx.x & 31;
  const int half = lane >> 4;
  const int rc   = lane & 15;
  const bf16_t* Ar = A + rc * lda;
  const bf16_t* Bc = B + rc * ldb;
  v8f c0 = c;
  v8f c1 = {};
  int k = 0;
  for (; k + 64 <= K; k += 64) {
    const int ka = k + half * 8;
    FragU a0, b0, a1, b1;
    a0.u[0] = *(const uint4*)(Ar + ka);
    a0.u[1] = *(const uint4*)(Ar + ka + 16);
    b0.u[0] = *(const uint4*)(Bc + ka);
    b0.u[1] = *(const uint4*)(Bc + ka + 16);
    a1.u[0] = *(const uint4*)(Ar + ka + 32);
    a1.u[1] = *(const uint4*)(Ar + ka + 48);
    b1.u[0] = *(const uint4*)(Bc + ka + 32);
    b1.u[1] = *(const uint4*)(Bc + ka + 48);
    c0 = __builtin_amdgcn_wmma_f32_16x16x32_bf16(false, a0.v, false, b0.v,
                                                 (short)0, c0, false, false);
    c1 = __builtin_amdgcn_wmma_f32_16x16x32_bf16(false, a1.v, false, b1.v,
                                                 (short)0, c1, false, false);
  }
  if (k < K) {                 // K % 64 == 32 remainder (e.g. K = 32)
    const int ka = k + half * 8;
    FragU a0, b0;
    a0.u[0] = *(const uint4*)(Ar + ka);
    a0.u[1] = *(const uint4*)(Ar + ka + 16);
    b0.u[0] = *(const uint4*)(Bc + ka);
    b0.u[1] = *(const uint4*)(Bc + ka + 16);
    c0 = __builtin_amdgcn_wmma_f32_16x16x32_bf16(false, a0.v, false, b0.v,
                                                 (short)0, c0, false, false);
  }
  return c0 + c1;
}

// ---------------- weight prep: fp32 -> bf16 column-major (B operand) --------
// dst[n*Kpad + k] = (k<K && n<N) ? src[k*N + n] : 0     (transpose + cast)
__global__ void transpose_cast_pad(const float* __restrict__ src,
                                   bf16_t* __restrict__ dst,
                                   int K, int N, int Kpad, int Npad) {
  int i = blockIdx.x * blockDim.x + threadIdx.x;
  if (i >= Npad * Kpad) return;
  int n = i / Kpad, k = i % Kpad;
  float v = (k < K && n < N) ? src[k * N + n] : 0.0f;
  dst[i] = (bf16_t)v;
}

// dst[n*Kpad + k] = (k<K) ? src[n*K + k] : 0            (cast, K pad)
// (col-major form of W^T == row-major W)
__global__ void cast_pad(const float* __restrict__ src,
                         bf16_t* __restrict__ dst,
                         int K, int Kpad, int N) {
  int i = blockIdx.x * blockDim.x + threadIdx.x;
  if (i >= N * Kpad) return;
  int n = i / Kpad, k = i % Kpad;
  dst[i] = (bf16_t)((k < K) ? src[n * K + k] : 0.0f);
}

// ---------------- x1_pred = sigmoid(x @ W1 + b1), once --------------------
__global__ void __launch_bounds__(256)
x1pred_kernel(const float* __restrict__ x,      // [B][784] fp32
              const bf16_t* __restrict__ W1cm,  // [256][800] col-major bf16
              const float* __restrict__ b1,
              bf16_t* __restrict__ x1pred) {    // [B][256] bf16
  const int lane = threadIdx.x & 31;
  const int w    = threadIdx.x >> 5;
  const int half = lane >> 4;
  const int rc   = lane & 15;
  const int tile = blockIdx.x * 8 + w;          // (B/16)*(D1/16) tiles
  const int mt   = tile / (PCN_D1 / 16);
  const int nt   = tile % (PCN_D1 / 16);

  const float*  Ar = x + (mt * 16 + rc) * PCN_DIN;
  const bf16_t* Bc = W1cm + (nt * 16 + rc) * PCN_DINP;

  v8f c = {};
  for (int k = 0; k < PCN_DINP; k += 32) {
    const int ka = k + half * 8;
    FragU a, b;
#pragma unroll
    for (int run = 0; run < 2; ++run) {
      const int kb = ka + run * 16;
      float fv[8];
      if (kb < PCN_DIN) {               // 784 is a multiple of 8: run all-in/out
        const float4 t0 = *(const float4*)(Ar + kb);
        const float4 t1 = *(const float4*)(Ar + kb + 4);
        fv[0]=t0.x; fv[1]=t0.y; fv[2]=t0.z; fv[3]=t0.w;
        fv[4]=t1.x; fv[5]=t1.y; fv[6]=t1.z; fv[7]=t1.w;
      } else {
#pragma unroll
        for (int j = 0; j < 8; ++j) fv[j] = 0.0f;
      }
#pragma unroll
      for (int j = 0; j < 8; ++j) a.e[run * 8 + j] = (bf16_t)fv[j];
      b.u[run] = *(const uint4*)(Bc + kb);
    }
    c = __builtin_amdgcn_wmma_f32_16x16x32_bf16(false, a.v, false, b.v,
                                                (short)0, c, false, false);
  }
#pragma unroll
  for (int i = 0; i < 8; ++i) {
    const int r   = mt * 16 + half * 8 + i;
    const int col = nt * 16 + rc;
    x1pred[r * PCN_D1 + col] = (bf16_t)sigmoidf_(c[i] + b1[col]);
  }
}

// ---------------- fused 20-step PCN inference, one block = 32 rows ----------
#define SMEM_ELEMS (2*MT*LD1 + 2*MT*LD2 + 2*MT*LD3 + MT*LD2 + MT*LD3 + MT*LD4)
#define SMEM_BYTES (SMEM_ELEMS * 2)   // 285184 B < 320 KB/WGP

__global__ void __launch_bounds__(256)
pcn_step_kernel(const float* __restrict__ x1_0,
                const float* __restrict__ x2_0,
                const float* __restrict__ x3_0,
                const float* __restrict__ x4_0,   // [B][10]
                const float* __restrict__ b2,
                const float* __restrict__ b3,
                const float* __restrict__ b4,
                const bf16_t* __restrict__ W2cm,  // [1024][256]
                const bf16_t* __restrict__ W2rm,  // [256][1024]  (= W2^T col-major)
                const bf16_t* __restrict__ W3cm,  // [256][1024]
                const bf16_t* __restrict__ W3rm,  // [1024][256]
                const bf16_t* __restrict__ W4cm,  // [16][256]   (N padded)
                const bf16_t* __restrict__ W4Tcm, // [256][32]   (K padded)
                const bf16_t* __restrict__ L1cm,  // [256][256]
                const bf16_t* __restrict__ L2cm,  // [1024][1024]
                const bf16_t* __restrict__ L3cm,  // [256][256]
                const bf16_t* __restrict__ x1pred,// [B][256]
                float* __restrict__ out) {        // [B][10]
  extern __shared__ bf16_t smem[];
  bf16_t* s_x1 = smem;                        // 2 * 32*LD1
  bf16_t* s_x2 = s_x1 + 2 * MT * LD1;         // 2 * 32*LD2
  bf16_t* s_x3 = s_x2 + 2 * MT * LD2;         // 2 * 32*LD3
  bf16_t* s_e2 = s_x3 + 2 * MT * LD3;         // 32*LD2
  bf16_t* s_e3 = s_e2 + MT * LD2;             // 32*LD3
  bf16_t* s_e4 = s_e3 + MT * LD3;             // 32*LD4 (K padded to 32)

  const int tid  = threadIdx.x;
  const int lane = tid & 31;
  const int w    = tid >> 5;      // 0..7
  const int wm   = w >> 2;        // row half of the 32-row tile
  const int wn   = w & 3;         // column group
  const int half = lane >> 4;
  const int rc   = lane & 15;
  const int row0 = blockIdx.x * MT;

  // ---- init: states fp32->bf16 into LDS (padded rows), zero eps4 buffer
  for (int e = tid; e < MT * PCN_D1; e += 256) {
    int r = e / PCN_D1, cc = e % PCN_D1;
    s_x1[r * LD1 + cc] = (bf16_t)x1_0[row0 * PCN_D1 + e];
  }
  for (int e = tid; e < MT * PCN_D2; e += 256) {
    int r = e / PCN_D2, cc = e % PCN_D2;
    s_x2[r * LD2 + cc] = (bf16_t)x2_0[row0 * PCN_D2 + e];
  }
  for (int e = tid; e < MT * PCN_D3; e += 256) {
    int r = e / PCN_D3, cc = e % PCN_D3;
    s_x3[r * LD3 + cc] = (bf16_t)x3_0[row0 * PCN_D3 + e];
  }
  for (int e = tid; e < MT * LD4; e += 256) s_e4[e] = (bf16_t)0.0f;
  __syncthreads();

  int cur = 0;
  for (int step = 0; step < PCN_STEPS; ++step) {
    bf16_t* x1c = s_x1 + cur * MT * LD1;
    bf16_t* x2c = s_x2 + cur * MT * LD2;
    bf16_t* x3c = s_x3 + cur * MT * LD3;
    bf16_t* x1n = s_x1 + (cur ^ 1) * MT * LD1;
    bf16_t* x2n = s_x2 + (cur ^ 1) * MT * LD2;
    bf16_t* x3n = s_x3 + (cur ^ 1) * MT * LD3;

    // P1: eps3 = sigmoid(x2 @ W3 + b3) - x3      [32x256], K=1024
    for (int nt = wn; nt < PCN_D3 / 16; nt += 4) {
      v8f c = {};
      c = wmma_AB(x2c + wm * 16 * LD2, LD2,
                  W3cm + nt * 16 * PCN_D2, PCN_D2, PCN_D2, c);
#pragma unroll
      for (int i = 0; i < 8; ++i) {
        const int r   = wm * 16 + half * 8 + i;
        const int col = nt * 16 + rc;
        const float v = sigmoidf_(c[i] + b3[col]) - (float)x3c[r * LD3 + col];
        s_e3[r * LD3 + col] = (bf16_t)v;
      }
    }
    // P2: eps2 = sigmoid(x1 @ W2 + b2) - x2      [32x1024], K=256
    for (int nt = wn; nt < PCN_D2 / 16; nt += 4) {
      v8f c = {};
      c = wmma_AB(x1c + wm * 16 * LD1, LD1,
                  W2cm + nt * 16 * PCN_D1, PCN_D1, PCN_D1, c);
#pragma unroll
      for (int i = 0; i < 8; ++i) {
        const int r   = wm * 16 + half * 8 + i;
        const int col = nt * 16 + rc;
        const float v = sigmoidf_(c[i] + b2[col]) - (float)x2c[r * LD2 + col];
        s_e2[r * LD2 + col] = (bf16_t)v;
      }
    }
    // P3: eps4 = (x3 @ W4 + b4) - x4_0           [32x16->pad32], K=256
    if (wn == 0) {
      v8f c = {};
      c = wmma_AB(x3c + wm * 16 * LD3, LD3, W4cm, PCN_D3, PCN_D3, c);
#pragma unroll
      for (int i = 0; i < 8; ++i) {
        const int r   = wm * 16 + half * 8 + i;
        const int col = rc;
        float v = 0.0f;
        if (col < PCN_DOUT)
          v = (c[i] + b4[col]) - x4_0[(row0 + r) * PCN_DOUT + col];
        s_e4[r * LD4 + col] = (bf16_t)v;
      }
    }
    __syncthreads();

    // P4: x3' = x3 - LR*(-eps3 + eps4@W4^T - g*x3@L3)   K1=32, K2=256
    for (int nt = wn; nt < PCN_D3 / 16; nt += 4) {
      v8f c1 = {}, c2 = {};
      c1 = wmma_AB(s_e4 + wm * 16 * LD4, LD4, W4Tcm + nt * 16 * 32, 32, 32, c1);
      c2 = wmma_AB(x3c + wm * 16 * LD3, LD3,
                   L3cm + nt * 16 * PCN_D3, PCN_D3, PCN_D3, c2);
#pragma unroll
      for (int i = 0; i < 8; ++i) {
        const int r   = wm * 16 + half * 8 + i;
        const int col = nt * 16 + rc;
        const float xv = (float)x3c[r * LD3 + col];
        const float g  = -(float)s_e3[r * LD3 + col] + c1[i] - PCN_GAMMA * c2[i];
        x3n[r * LD3 + col] = (bf16_t)(xv - PCN_LR * g);
      }
    }
    // P5: x2' = x2 - LR*(-eps2 + eps3@W3^T - g*x2@L2)   K1=256, K2=1024
    for (int nt = wn; nt < PCN_D2 / 16; nt += 4) {
      v8f c1 = {}, c2 = {};
      c1 = wmma_AB(s_e3 + wm * 16 * LD3, LD3,
                   W3rm + nt * 16 * PCN_D3, PCN_D3, PCN_D3, c1);
      c2 = wmma_AB(x2c + wm * 16 * LD2, LD2,
                   L2cm + nt * 16 * PCN_D2, PCN_D2, PCN_D2, c2);
#pragma unroll
      for (int i = 0; i < 8; ++i) {
        const int r   = wm * 16 + half * 8 + i;
        const int col = nt * 16 + rc;
        const float xv = (float)x2c[r * LD2 + col];
        const float g  = -(float)s_e2[r * LD2 + col] + c1[i] - PCN_GAMMA * c2[i];
        x2n[r * LD2 + col] = (bf16_t)(xv - PCN_LR * g);
      }
    }
    // P6: x1' = x1 - LR*(-(x1pred-x1) + eps2@W2^T - g*x1@L1)  K1=1024, K2=256
    for (int nt = wn; nt < PCN_D1 / 16; nt += 4) {
      v8f c1 = {}, c2 = {};
      c1 = wmma_AB(s_e2 + wm * 16 * LD2, LD2,
                   W2rm + nt * 16 * PCN_D2, PCN_D2, PCN_D2, c1);
      c2 = wmma_AB(x1c + wm * 16 * LD1, LD1,
                   L1cm + nt * 16 * PCN_D1, PCN_D1, PCN_D1, c2);
#pragma unroll
      for (int i = 0; i < 8; ++i) {
        const int r   = wm * 16 + half * 8 + i;
        const int col = nt * 16 + rc;
        const float xv = (float)x1c[r * LD1 + col];
        const float e1 = (float)x1pred[(row0 + r) * PCN_D1 + col] - xv;
        const float g  = -e1 + c1[i] - PCN_GAMMA * c2[i];
        x1n[r * LD1 + col] = (bf16_t)(xv - PCN_LR * g);
      }
    }
    __syncthreads();
    cur ^= 1;
  }

  // final: out = x3 @ W4 + b4 (only the 10 real columns)
  const bf16_t* x3f = s_x3 + cur * MT * LD3;
  if (wn == 0) {
    v8f c = {};
    c = wmma_AB(x3f + wm * 16 * LD3, LD3, W4cm, PCN_D3, PCN_D3, c);
#pragma unroll
    for (int i = 0; i < 8; ++i) {
      const int r   = wm * 16 + half * 8 + i;
      const int col = rc;
      if (col < PCN_DOUT)
        out[(row0 + r) * PCN_DOUT + col] = c[i] + b4[col];
    }
  }
}

// ---------------------------------------------------------------------------
extern "C" void kernel_launch(void* const* d_in, const int* in_sizes, int n_in,
                              void* d_out, int out_size, void* d_ws, size_t ws_size,
                              hipStream_t stream) {
  (void)in_sizes; (void)n_in; (void)out_size; (void)ws_size;
  const float* x    = (const float*)d_in[0];
  const float* x1_0 = (const float*)d_in[1];
  const float* x2_0 = (const float*)d_in[2];
  const float* x3_0 = (const float*)d_in[3];
  const float* x4_0 = (const float*)d_in[4];
  const float* W1   = (const float*)d_in[5];
  const float* W2   = (const float*)d_in[6];
  const float* W3   = (const float*)d_in[7];
  const float* W4   = (const float*)d_in[8];
  const float* b1   = (const float*)d_in[9];
  const float* b2   = (const float*)d_in[10];
  const float* b3   = (const float*)d_in[11];
  const float* b4   = (const float*)d_in[12];
  const float* L1   = (const float*)d_in[13];
  const float* L2   = (const float*)d_in[14];
  const float* L3   = (const float*)d_in[15];
  float* out = (float*)d_out;

  // workspace carve-out (bf16 weight copies + x1_pred), 256B aligned
  char* ws = (char*)d_ws;
  size_t off = 0;
  auto carve = [&](size_t elems) {
    bf16_t* p = (bf16_t*)(ws + off);
    off += (elems * sizeof(bf16_t) + 255) & ~(size_t)255;
    return p;
  };
  bf16_t* W1cm   = carve((size_t)PCN_D1 * PCN_DINP);   // [256][800]
  bf16_t* W2cm   = carve((size_t)PCN_D2 * PCN_D1);     // [1024][256]
  bf16_t* W2rm   = carve((size_t)PCN_D1 * PCN_D2);     // [256][1024]
  bf16_t* W3cm   = carve((size_t)PCN_D3 * PCN_D2);     // [256][1024]
  bf16_t* W3rm   = carve((size_t)PCN_D2 * PCN_D3);     // [1024][256]
  bf16_t* W4cm   = carve((size_t)16 * PCN_D3);         // [16][256]
  bf16_t* W4Tcm  = carve((size_t)PCN_D3 * 32);         // [256][32]
  bf16_t* L1cm   = carve((size_t)PCN_D1 * PCN_D1);
  bf16_t* L2cm   = carve((size_t)PCN_D2 * PCN_D2);
  bf16_t* L3cm   = carve((size_t)PCN_D3 * PCN_D3);
  bf16_t* x1pred = carve((size_t)PCN_B * PCN_D1);      // [8192][256]

  const int TB = 256;
  auto grid = [&](int total) { return (total + TB - 1) / TB; };

  // B-operand prep: col-major bf16 (transpose) or row-major bf16 (for W^T)
  transpose_cast_pad<<<grid(PCN_D1 * PCN_DINP), TB, 0, stream>>>(W1, W1cm, PCN_DIN, PCN_D1, PCN_DINP, PCN_D1);
  transpose_cast_pad<<<grid(PCN_D2 * PCN_D1),  TB, 0, stream>>>(W2, W2cm, PCN_D1, PCN_D2, PCN_D1, PCN_D2);
  cast_pad          <<<grid(PCN_D1 * PCN_D2),  TB, 0, stream>>>(W2, W2rm, PCN_D2, PCN_D2, PCN_D1);
  transpose_cast_pad<<<grid(PCN_D3 * PCN_D2),  TB, 0, stream>>>(W3, W3cm, PCN_D2, PCN_D3, PCN_D2, PCN_D3);
  cast_pad          <<<grid(PCN_D2 * PCN_D3),  TB, 0, stream>>>(W3, W3rm, PCN_D3, PCN_D3, PCN_D2);
  transpose_cast_pad<<<grid(16 * PCN_D3),      TB, 0, stream>>>(W4, W4cm, PCN_D3, PCN_DOUT, PCN_D3, 16);
  cast_pad          <<<grid(PCN_D3 * 32),      TB, 0, stream>>>(W4, W4Tcm, PCN_DOUT, 32, PCN_D3);
  transpose_cast_pad<<<grid(PCN_D1 * PCN_D1),  TB, 0, stream>>>(L1, L1cm, PCN_D1, PCN_D1, PCN_D1, PCN_D1);
  transpose_cast_pad<<<grid(PCN_D2 * PCN_D2),  TB, 0, stream>>>(L2, L2cm, PCN_D2, PCN_D2, PCN_D2, PCN_D2);
  transpose_cast_pad<<<grid(PCN_D3 * PCN_D3),  TB, 0, stream>>>(L3, L3cm, PCN_D3, PCN_D3, PCN_D3, PCN_D3);

  // x1_pred (loop-invariant forward layer), WMMA bf16
  {
    const int tiles = (PCN_B / 16) * (PCN_D1 / 16);   // 8192 tiles, 8 per block
    x1pred_kernel<<<tiles / 8, TB, 0, stream>>>(x, W1cm, b1, x1pred);
  }

  // fused persistent 20-step kernel: 256 blocks x 256 threads, ~279 KB LDS each
  (void)hipFuncSetAttribute((const void*)pcn_step_kernel,
                            hipFuncAttributeMaxDynamicSharedMemorySize,
                            SMEM_BYTES);
  pcn_step_kernel<<<PCN_B / MT, TB, SMEM_BYTES, stream>>>(
      x1_0, x2_0, x3_0, x4_0, b2, b3, b4,
      W2cm, W2rm, W3cm, W3rm, W4cm, W4Tcm, L1cm, L2cm, L3cm,
      x1pred, out);
}
